// lstm_autoencoder_32074815767367
// MI455X (gfx1250) — matrix operand order repaired and checked by reading, not compiled, not run on hardware
//
#include <hip/hip_runtime.h>
#include <stdint.h>

#define B_   128
#define T_   512
#define I_   128
#define H_   1024
#define NBLK 64
#define NTHR 256

#define WHH_STRIDE 1032   // 1024 + 8 bf16 pad -> conflict-free ds_load_b128
#define WIH_STRIDE 136    // 128 + 8 bf16 pad

typedef __attribute__((ext_vector_type(16))) __bf16 v16bf;
typedef __attribute__((ext_vector_type(8)))  float  v8f;

union FragAB {
    uint4          u4[2];
    unsigned short us[16];
    v16bf          v;
};

__device__ __forceinline__ unsigned short f2bf(float f) {
    unsigned u = __float_as_uint(f);
    unsigned r = u + 0x7fffu + ((u >> 16) & 1u);   // round-to-nearest-even
    return (unsigned short)(r >> 16);
}

__device__ __forceinline__ v8f vzero8() {
    v8f z;
#pragma unroll
    for (int e = 0; e < 8; ++e) z[e] = 0.0f;
    return z;
}

__device__ __forceinline__ v8f wmma_bf16(v16bf a, v16bf b, v8f c) {
    return __builtin_amdgcn_wmma_f32_16x16x32_bf16(false, a, false, b, (short)0, c,
                                                   false, false);
}

__device__ __forceinline__ float sigf(float x) { return 1.0f / (1.0f + __expf(-x)); }

// ---------------- grid-wide software barrier (persistent kernels) -------------
__device__ __forceinline__ void grid_barrier(unsigned* ctr, unsigned* gen, unsigned nb) {
    __syncthreads();
    if (threadIdx.x == 0) {
        __threadfence();
        unsigned g = __atomic_load_n(gen, __ATOMIC_ACQUIRE);
        if (atomicAdd(ctr, 1u) == nb - 1u) {
            __atomic_store_n(ctr, 0u, __ATOMIC_RELAXED);
            __atomic_fetch_add(gen, 1u, __ATOMIC_RELEASE);
        } else {
            while (__atomic_load_n(gen, __ATOMIC_ACQUIRE) == g)
                __builtin_amdgcn_s_sleep(2);
        }
    }
    __syncthreads();
}

// ---- stage this block's weight slices (Whh 128KB, Wih 16KB) into LDS --------
__device__ __forceinline__ void stage_weights(const unsigned short* __restrict__ Whh,
                                              const unsigned short* __restrict__ Wih,
                                              unsigned short* ldsWhh,
                                              unsigned short* ldsWih, int blk) {
    // Whh: 64 local rows (g*16+jl -> global row g*1024 + blk*16 + jl), 1024 K each
    for (int u = threadIdx.x; u < 64 * (H_ / 8); u += NTHR) {
        const int rl = u >> 7;            // 128 uint4 per row
        const int c8 = u & 127;
        const int g = rl >> 4, jl = rl & 15;
        const uint4* src =
            (const uint4*)(Whh + (size_t)(g * H_ + blk * 16 + jl) * H_) + c8;
        *(uint4*)(ldsWhh + rl * WHH_STRIDE + c8 * 8) = *src;
    }
    // Wih: 64 local rows, 128 K each
    for (int u = threadIdx.x; u < 64 * (I_ / 8); u += NTHR) {
        const int rl = u >> 4;            // 16 uint4 per row
        const int c8 = u & 15;
        const int g = rl >> 4, jl = rl & 15;
        const uint4* src =
            (const uint4*)(Wih + (size_t)(g * H_ + blk * 16 + jl) * I_) + c8;
        *(uint4*)(ldsWih + rl * WIH_STRIDE + c8 * 8) = *src;
    }
    __syncthreads();
}

// ---------------- prologue kernels -------------------------------------------
__global__ void cvt_bf16_kernel(const float* __restrict__ src,
                                unsigned short* __restrict__ dst, int n) {
    int i = blockIdx.x * blockDim.x + threadIdx.x;
    if (i < n) dst[i] = f2bf(src[i]);
}

__global__ void add_bias_kernel(const float* __restrict__ a, const float* __restrict__ b,
                                float* __restrict__ d, int n) {
    int i = blockIdx.x * blockDim.x + threadIdx.x;
    if (i < n) d[i] = a[i] + b[i];
}

__global__ void init_xlast_kernel(const float* __restrict__ x,
                                  unsigned short* __restrict__ xbf) {
    int i = blockIdx.x * blockDim.x + threadIdx.x;
    if (i < B_ * I_) {
        int bb = i / I_, col = i % I_;
        xbf[i] = f2bf(x[((size_t)bb * T_ + (T_ - 1)) * I_ + col]);
    }
}

// ---------------- encoder scan (persistent, 1 barrier / step) ----------------
__global__ __launch_bounds__(NTHR, 1)
void enc_scan(const unsigned short* __restrict__ x_bf16, // (B,T,I) bf16, prepacked
              const unsigned short* __restrict__ Whh,    // (4096,1024) bf16 row-major
              const unsigned short* __restrict__ Wih,    // (4096,128)  bf16 row-major
              const float* __restrict__ bias,            // 4096, bih+bhh
              const int* __restrict__ lengths,
              unsigned short* __restrict__ h_bf,         // (128,1024) bf16
              float* __restrict__ c_ws,                  // (128,1024) f32 (final c)
              float* __restrict__ enc_out,               // (B,T,H) f32
              unsigned* __restrict__ bar) {
    __shared__ unsigned short ldsWhh[64 * WHH_STRIDE];   // 132096 B
    __shared__ unsigned short ldsWih[64 * WIH_STRIDE];   //  17408 B

    const int blk  = blockIdx.x;           // 0..63, owns j0 = 16*blk
    const int wave = threadIdx.x >> 5;     // 0..7, owns rows 16*wave..+15
    const int lane = threadIdx.x & 31;
    const int half = lane >> 4;
    const int l16  = lane & 15;
    const int j    = blk * 16 + l16;       // hidden unit (C/B column)
    const int arow = wave * 16 + l16;      // A-fragment row (batch index)
    const int koff = half * 16;            // A per-lane K offset within chunk
    const int rowb = wave * 16 + half * 8; // C-tile row base for this lane

    stage_weights(Whh, Wih, ldsWhh, ldsWih, blk);

    float bg[4];
#pragma unroll
    for (int g = 0; g < 4; ++g) bg[g] = bias[g * H_ + j];

    int len[8];
#pragma unroll
    for (int r = 0; r < 8; ++r) len[r] = lengths[rowb + r];

    float creg[8];
#pragma unroll
    for (int r = 0; r < 8; ++r) creg[r] = 0.0f;

    const unsigned short* Ah     = h_bf + (size_t)arow * H_;
    const unsigned short* Axbase = x_bf16 + (size_t)arow * (T_ * I_);

    for (int t = 0; t < T_; ++t) {
        v8f acc[4];
#pragma unroll
        for (int g = 0; g < 4; ++g) acc[g] = vzero8();

        // ---- h @ Whh^T : K = 1024, B from LDS --------------------------------
#pragma unroll 2
        for (int kc = 0; kc < H_ / 32; ++kc) {
            const int k0 = kc * 32;
            FragAB a;
            a.u4[0] = *(const uint4*)(Ah + k0 + koff);
            a.u4[1] = *(const uint4*)(Ah + k0 + koff + 8);
#pragma unroll
            for (int g = 0; g < 4; ++g) {
                const unsigned short* bp =
                    ldsWhh + (g * 16 + l16) * WHH_STRIDE + k0 + half * 16;
                FragAB b;
                b.u4[0] = *(const uint4*)(bp);
                b.u4[1] = *(const uint4*)(bp + 8);
                acc[g] = wmma_bf16(a.v, b.v, acc[g]);
            }
        }

        // ---- x_t @ Wih^T : K = 128, B from LDS -------------------------------
        const unsigned short* Ax = Axbase + (size_t)t * I_;
#pragma unroll
        for (int kc = 0; kc < I_ / 32; ++kc) {
            const int k0 = kc * 32;
            FragAB a;
            a.u4[0] = *(const uint4*)(Ax + k0 + koff);
            a.u4[1] = *(const uint4*)(Ax + k0 + koff + 8);
#pragma unroll
            for (int g = 0; g < 4; ++g) {
                const unsigned short* bp =
                    ldsWih + (g * 16 + l16) * WIH_STRIDE + k0 + half * 16;
                FragAB b;
                b.u4[0] = *(const uint4*)(bp);
                b.u4[1] = *(const uint4*)(bp + 8);
                acc[g] = wmma_bf16(a.v, b.v, acc[g]);
            }
        }

        // ---- gates + state update (all in-lane, c lives in registers) -------
#pragma unroll
        for (int r = 0; r < 8; ++r) {
            float iv = sigf(acc[0][r] + bg[0]);
            float fv = sigf(acc[1][r] + bg[1]);
            float gv = tanhf(acc[2][r] + bg[2]);
            float ov = sigf(acc[3][r] + bg[3]);
            float c2 = fv * creg[r] + iv * gv;
            float h2 = ov * tanhf(c2);
            const int  b = rowb + r;
            const bool v = (t < len[r]);
            if (v) {
                creg[r]                  = c2;
                h_bf[(size_t)b * H_ + j] = f2bf(h2);   // masked lanes keep old h
            }
            enc_out[((size_t)b * T_ + t) * H_ + j] = v ? h2 : 0.0f;
        }

        __threadfence();
        grid_barrier(bar, bar + 1, NBLK);
    }

#pragma unroll
    for (int r = 0; r < 8; ++r)
        c_ws[(size_t)(rowb + r) * H_ + j] = creg[r];
}

// ---------------- decoder scan (persistent, 2 barriers / step) ---------------
__global__ __launch_bounds__(NTHR, 1)
void dec_scan(const unsigned short* __restrict__ x_bf16, // (B,T,I) bf16 (teacher)
              const unsigned short* __restrict__ Whh,
              const unsigned short* __restrict__ Wih,
              const unsigned short* __restrict__ Wlin,   // (128,1024) bf16 row-major
              const float* __restrict__ bias,
              const float* __restrict__ lin_b,
              const int* __restrict__ tf_mask,
              unsigned short* __restrict__ h_bf,
              unsigned short* __restrict__ x_bf,         // (128,128) bf16 next input
              const float* __restrict__ c_ws,
              float* __restrict__ outputs,               // (B,T,I) f32
              unsigned* __restrict__ bar) {
    __shared__ unsigned short ldsWhh[64 * WHH_STRIDE];
    __shared__ unsigned short ldsWih[64 * WIH_STRIDE];

    const int blk  = blockIdx.x;
    const int wave = threadIdx.x >> 5;
    const int lane = threadIdx.x & 31;
    const int half = lane >> 4;
    const int l16  = lane & 15;
    const int j    = blk * 16 + l16;
    const int arow = wave * 16 + l16;
    const int koff = half * 16;
    const int rowb = wave * 16 + half * 8;

    stage_weights(Whh, Wih, ldsWhh, ldsWih, blk);

    float bg[4];
#pragma unroll
    for (int g = 0; g < 4; ++g) bg[g] = bias[g * H_ + j];

    float creg[8];
#pragma unroll
    for (int r = 0; r < 8; ++r) creg[r] = c_ws[(size_t)(rowb + r) * H_ + j];

    const unsigned short* Ah = h_bf + (size_t)arow * H_;
    const unsigned short* Ax = x_bf + (size_t)arow * I_;
    const int   n  = blk * 16 + l16;                       // projection column
    const float lb = (blk < 8) ? lin_b[n] : 0.0f;

    for (int t = 0; t < T_; ++t) {
        v8f acc[4];
#pragma unroll
        for (int g = 0; g < 4; ++g) acc[g] = vzero8();

        // ---- h @ Whh^T, B from LDS ------------------------------------------
#pragma unroll 2
        for (int kc = 0; kc < H_ / 32; ++kc) {
            const int k0 = kc * 32;
            FragAB a;
            a.u4[0] = *(const uint4*)(Ah + k0 + koff);
            a.u4[1] = *(const uint4*)(Ah + k0 + koff + 8);
#pragma unroll
            for (int g = 0; g < 4; ++g) {
                const unsigned short* bp =
                    ldsWhh + (g * 16 + l16) * WHH_STRIDE + k0 + half * 16;
                FragAB b;
                b.u4[0] = *(const uint4*)(bp);
                b.u4[1] = *(const uint4*)(bp + 8);
                acc[g] = wmma_bf16(a.v, b.v, acc[g]);
            }
        }

        // ---- x_in @ Wih^T, B from LDS ---------------------------------------
#pragma unroll
        for (int kc = 0; kc < I_ / 32; ++kc) {
            const int k0 = kc * 32;
            FragAB a;
            a.u4[0] = *(const uint4*)(Ax + k0 + koff);
            a.u4[1] = *(const uint4*)(Ax + k0 + koff + 8);
#pragma unroll
            for (int g = 0; g < 4; ++g) {
                const unsigned short* bp =
                    ldsWih + (g * 16 + l16) * WIH_STRIDE + k0 + half * 16;
                FragAB b;
                b.u4[0] = *(const uint4*)(bp);
                b.u4[1] = *(const uint4*)(bp + 8);
                acc[g] = wmma_bf16(a.v, b.v, acc[g]);
            }
        }

        // ---- gates (no masking in decoder) -----------------------------------
#pragma unroll
        for (int r = 0; r < 8; ++r) {
            float iv = sigf(acc[0][r] + bg[0]);
            float fv = sigf(acc[1][r] + bg[1]);
            float gv = tanhf(acc[2][r] + bg[2]);
            float ov = sigf(acc[3][r] + bg[3]);
            float c2 = fv * creg[r] + iv * gv;
            creg[r]  = c2;
            float h2 = ov * tanhf(c2);
            h_bf[(size_t)(rowb + r) * H_ + j] = f2bf(h2);
        }

        __threadfence();
        grid_barrier(bar, bar + 1, NBLK);   // h complete

        // ---- out = h @ lin_W^T + lin_b ; select next input -------------------
        if (blk < 8) {
            v8f po = vzero8();
#pragma unroll 2
            for (int kc = 0; kc < H_ / 32; ++kc) {
                const int k0 = kc * 32;
                FragAB a;
                a.u4[0] = *(const uint4*)(Ah + k0 + koff);
                a.u4[1] = *(const uint4*)(Ah + k0 + koff + 8);
                const unsigned short* bp = Wlin + (size_t)n * H_ + k0 + half * 16;
                FragAB b;
                b.u4[0] = *(const uint4*)(bp);
                b.u4[1] = *(const uint4*)(bp + 8);
                po = wmma_bf16(a.v, b.v, po);
            }
            const bool tf = (tf_mask[t] != 0);
#pragma unroll
            for (int r = 0; r < 8; ++r) {
                const int b   = rowb + r;
                float     val = po[r] + lb;
                outputs[((size_t)b * T_ + t) * I_ + n] = val;
                unsigned short xi =
                    tf ? x_bf16[((size_t)b * T_ + t) * I_ + n] : f2bf(val);
                x_bf[(size_t)b * I_ + n] = xi;
            }
        }

        __threadfence();
        grid_barrier(bar, bar + 1, NBLK);   // out / x_in complete
    }
}

// ---------------- host side ---------------------------------------------------
extern "C" void kernel_launch(void* const* d_in, const int* in_sizes, int n_in,
                              void* d_out, int out_size, void* d_ws, size_t ws_size,
                              hipStream_t stream) {
    const float* x_in    = (const float*)d_in[0];
    const float* encWih  = (const float*)d_in[1];
    const float* encWhh  = (const float*)d_in[2];
    const float* encBih  = (const float*)d_in[3];
    const float* encBhh  = (const float*)d_in[4];
    const float* decWih  = (const float*)d_in[5];
    const float* decWhh  = (const float*)d_in[6];
    const float* decBih  = (const float*)d_in[7];
    const float* decBhh  = (const float*)d_in[8];
    const float* linW    = (const float*)d_in[9];
    const float* linB    = (const float*)d_in[10];
    const int*   lengths = (const int*)d_in[11];
    const int*   tfmask  = (const int*)d_in[12];

    char*  ws = (char*)d_ws;
    size_t o  = 0;
    auto carve = [&](size_t bytes) {
        size_t r = o;
        o = (o + bytes + 255) & ~(size_t)255;
        return r;
    };
    unsigned short* encWih_bf = (unsigned short*)(ws + carve((size_t)4096 * 128 * 2));
    unsigned short* encWhh_bf = (unsigned short*)(ws + carve((size_t)4096 * 1024 * 2));
    unsigned short* decWih_bf = (unsigned short*)(ws + carve((size_t)4096 * 128 * 2));
    unsigned short* decWhh_bf = (unsigned short*)(ws + carve((size_t)4096 * 1024 * 2));
    unsigned short* linW_bf   = (unsigned short*)(ws + carve((size_t)128 * 1024 * 2));
    unsigned short* x_bf_all  = (unsigned short*)(ws + carve((size_t)B_ * T_ * I_ * 2));
    float*          encB      = (float*)(ws + carve((size_t)4096 * 4));
    float*          decB      = (float*)(ws + carve((size_t)4096 * 4));
    unsigned short* h_bf      = (unsigned short*)(ws + carve((size_t)B_ * H_ * 2));
    float*          c_ws      = (float*)(ws + carve((size_t)B_ * H_ * 4));
    unsigned short* x_bf      = (unsigned short*)(ws + carve((size_t)B_ * I_ * 2));
    unsigned*       bar       = (unsigned*)(ws + carve(256));

    float* outputs = (float*)d_out;                               // (B,T,I)
    float* enc_out = (float*)d_out + (size_t)B_ * T_ * I_;        // (B,T,H)

    // init: zero h0 (bf16 0x0000 == 0.0) and the barrier words
    hipMemsetAsync(h_bf, 0, (size_t)B_ * H_ * 2, stream);
    hipMemsetAsync(bar, 0, 256, stream);

    // weight / input conversion, bias fusion
    auto cvt = [&](const float* s, unsigned short* d, int n) {
        cvt_bf16_kernel<<<(n + 255) / 256, 256, 0, stream>>>(s, d, n);
    };
    cvt(encWih, encWih_bf, 4096 * 128);
    cvt(encWhh, encWhh_bf, 4096 * 1024);
    cvt(decWih, decWih_bf, 4096 * 128);
    cvt(decWhh, decWhh_bf, 4096 * 1024);
    cvt(linW,   linW_bf,   128 * 1024);
    cvt(x_in,   x_bf_all,  B_ * T_ * I_);
    add_bias_kernel<<<16, 256, 0, stream>>>(encBih, encBhh, encB, 4096);
    add_bias_kernel<<<16, 256, 0, stream>>>(decBih, decBhh, decB, 4096);
    init_xlast_kernel<<<(B_ * I_ + 255) / 256, 256, 0, stream>>>(x_in, x_bf);

    // encoder scan (persistent)
    enc_scan<<<NBLK, NTHR, 0, stream>>>(x_bf_all, encWhh_bf, encWih_bf, encB, lengths,
                                        h_bf, c_ws, enc_out, bar);

    // decoder scan (persistent)
    dec_scan<<<NBLK, NTHR, 0, stream>>>(x_bf_all, decWhh_bf, decWih_bf, linW_bf, decB,
                                        linB, tfmask, h_bf, x_bf, c_ws, outputs, bar);
}
